// PWQ_MultiHead_SelfAttention_28269474742415
// MI455X (gfx1250) — matrix operand order repaired and checked by reading, not compiled, not run on hardware
//
#include <hip/hip_runtime.h>
#include <hip/hip_bf16.h>

// ---------------------------------------------------------------------------
// MI455X (gfx1250) multi-head self-attention with nROPE, bf16 WMMA everywhere.
// B=4, T=1024, E=1024, H=16, HD=64, half=32.
// Final: async-to-LDS double-buffered staging, asm permlane16 max butterfly
// lane-partial exp sums, 1/sqrt(HD) folded into Q.
// ---------------------------------------------------------------------------

typedef __attribute__((ext_vector_type(16))) __bf16 v16bf;
typedef __attribute__((ext_vector_type(8)))  float  v8f;
typedef __attribute__((ext_vector_type(4)))  unsigned int u32x4;
typedef __attribute__((ext_vector_type(4)))  int i32x4;
typedef __attribute__((ext_vector_type(4)))  __bf16 bf16x4;

union Frag16 {            // one WMMA 16-element bf16 operand (8 VGPRs)
    v16bf v;
    u32x4 q[2];
};
union VReg {              // 8 bf16 staged through registers
    u32x4 q;
    __bf16 h[8];
};

#define WMMA_BF16(A, B, C) \
    __builtin_amdgcn_wmma_f32_16x16x32_bf16(false, (A), false, (B), (short)0, (C), false, false)

// ---- async global->LDS copy (CDNA5 ASYNCcnt path), guarded ----------------
#if __has_builtin(__builtin_amdgcn_global_load_async_to_lds_b128)
typedef __attribute__((address_space(1))) i32x4 GV4;   // global int4
typedef __attribute__((address_space(3))) i32x4 LV4;   // LDS int4
#define CP_ASYNC_B128(l, g) \
    __builtin_amdgcn_global_load_async_to_lds_b128((GV4*)(g), (LV4*)(l), 0, 0)
#if __has_builtin(__builtin_amdgcn_s_wait_asynccnt)
#define CP_WAIT() __builtin_amdgcn_s_wait_asynccnt(0)
#else
#define CP_WAIT() asm volatile("s_wait_asynccnt 0" ::: "memory")
#endif
#else
#define CP_ASYNC_B128(l, g) (*(u32x4*)(l) = *(const u32x4*)(g))
#define CP_WAIT() ((void)0)
#endif

// ---- raw VALU helpers (skip canonicalize / tied-old movs) -----------------
__device__ __forceinline__ float fmax_fast(float a, float b) {
    float d;
    asm("v_max_num_f32 %0, %1, %2" : "=v"(d) : "v"(a), "v"(b));
    return d;
}

// width-16 xor-butterfly max: 2 instructions per step
#define BFLY_MAX_STEP(v, t, lo, hi)                          \
    asm("v_permlane16_b32 %1, %0, %2, %3\n\t"                \
        "v_max_num_f32 %0, %0, %1"                           \
        : "+v"(v), "=&v"(t) : "s"(lo), "s"(hi))

__device__ __forceinline__ float red_max16_fast(float v) {
    float t;
    BFLY_MAX_STEP(v, t, 0x67452301u, 0xEFCDAB89u);   // xor 1
    BFLY_MAX_STEP(v, t, 0x54761032u, 0xDCFE98BAu);   // xor 2
    BFLY_MAX_STEP(v, t, 0x32107654u, 0xBA98FEDCu);   // xor 4
    BFLY_MAX_STEP(v, t, 0xFEDCBA98u, 0x76543210u);   // xor 8
    return v;
}

#define BFLY_ADD_STEP(v, t, lo, hi)                          \
    asm("v_permlane16_b32 %1, %0, %2, %3\n\t"                \
        "v_add_f32 %0, %0, %1"                               \
        : "+v"(v), "=&v"(t) : "s"(lo), "s"(hi))

__device__ __forceinline__ float red_sum16_fast(float v) {
    float t;
    BFLY_ADD_STEP(v, t, 0x67452301u, 0xEFCDAB89u);
    BFLY_ADD_STEP(v, t, 0x54761032u, 0xDCFE98BAu);
    BFLY_ADD_STEP(v, t, 0x32107654u, 0xBA98FEDCu);
    BFLY_ADD_STEP(v, t, 0xFEDCBA98u, 0x76543210u);
    return v;
}

// ---------------------------------------------------------------------------
// Pre-pass: x (f32) -> bf16, row-major [4096][1024]
// ---------------------------------------------------------------------------
__global__ void convert_x_kernel(const float* __restrict__ x, __bf16* __restrict__ xb) {
    int i = (blockIdx.x * 256 + threadIdx.x) * 4;
    float4 f = *(const float4*)(x + i);
    bf16x4 o;
    o[0] = (__bf16)f.x; o[1] = (__bf16)f.y; o[2] = (__bf16)f.z; o[3] = (__bf16)f.w;
    *(bf16x4*)(xb + i) = o;
}

// ---------------------------------------------------------------------------
// Pre-pass: W[k][n] (f32) -> Wt[n][k] (bf16), tiled transpose through LDS.
// ---------------------------------------------------------------------------
__global__ void transpose_w_kernel(const float* __restrict__ Wq, const float* __restrict__ Wk,
                                   const float* __restrict__ Wv, const float* __restrict__ Wo,
                                   __bf16* __restrict__ wqt, __bf16* __restrict__ wkt,
                                   __bf16* __restrict__ wvt, __bf16* __restrict__ wot) {
    __shared__ float tile[32][33];
    const float* W; __bf16* O;
    switch (blockIdx.z) {
        case 0:  W = Wq; O = wqt; break;
        case 1:  W = Wk; O = wkt; break;
        case 2:  W = Wv; O = wvt; break;
        default: W = Wo; O = wot; break;
    }
    int bx = blockIdx.x * 32, by = blockIdx.y * 32;
    int tx = threadIdx.x, ty = threadIdx.y;
    tile[ty][tx] = W[(by + ty) * 1024 + bx + tx];
    __syncthreads();
    O[(bx + ty) * 1024 + by + tx] = (__bf16)tile[tx][ty];
}

// ---------------------------------------------------------------------------
// bf16 WMMA GEMM, double-buffered async LDS staging.
//   A  : bf16 row-major [M=4096][K=1024]
//   Bt : bf16 row-major [N=1024][K=1024]
// MODE 0: bf16 scatter into per-head layout [B*H][T][64], (acc+bias)*oscale
// MODE 1: f32 row-major [M][N], acc+bias
// 256 threads = 8 waves; C-tile 128x128; wave tile 32x64.
// ---------------------------------------------------------------------------
template <int MODE>
__global__ __launch_bounds__(256, 1)
void gemm_bf16_kernel(const __bf16* __restrict__ A, const __bf16* __restrict__ Bt,
                      const float* __restrict__ bias, void* __restrict__ outp,
                      float oscale) {
    __shared__ __align__(16) __bf16 a_lds[2][128 * 40];
    __shared__ __align__(16) __bf16 b_lds[2][128 * 40];

    const int tid  = threadIdx.x;
    const int lane = tid & 31;
    const int w    = tid >> 5;
    const int hi   = lane >> 4;
    const int lm   = lane & 15;

    const int m0 = blockIdx.y * 128;
    const int n0 = blockIdx.x * 128;
    const int wm = (w >> 1) * 32;
    const int wn = (w & 1) * 64;

    // staging geometry: 512 x 16B segments per tile, 2 per thread per matrix
    const int srow0 = tid >> 2,          sc0 = (tid & 3) * 8;
    const int srow1 = (tid + 256) >> 2,  sc1 = sc0;   // (tid+256)&3 == tid&3

    v8f acc[2][4];
#pragma unroll
    for (int mt = 0; mt < 2; ++mt)
#pragma unroll
        for (int nt = 0; nt < 4; ++nt) acc[mt][nt] = {};

#define GEMM_STAGE(p, kc)                                                          \
    do {                                                                           \
        CP_ASYNC_B128(&a_lds[p][srow0 * 40 + sc0], &A[(m0 + srow0) * 1024 + (kc) + sc0]); \
        CP_ASYNC_B128(&b_lds[p][srow0 * 40 + sc0], &Bt[(n0 + srow0) * 1024 + (kc) + sc0]);\
        CP_ASYNC_B128(&a_lds[p][srow1 * 40 + sc1], &A[(m0 + srow1) * 1024 + (kc) + sc1]); \
        CP_ASYNC_B128(&b_lds[p][srow1 * 40 + sc1], &Bt[(n0 + srow1) * 1024 + (kc) + sc1]);\
    } while (0)

    GEMM_STAGE(0, 0);

    for (int it = 0; it < 32; ++it) {
        const int p = it & 1;
        CP_WAIT();
        __syncthreads();
        if (it + 1 < 32) GEMM_STAGE(1 - p, (it + 1) * 32);

        const __bf16* al = a_lds[p];
        const __bf16* bl = b_lds[p];
        Frag16 af[2], bf[4];
#pragma unroll
        for (int mt = 0; mt < 2; ++mt) {
            int base = (wm + mt * 16 + lm) * 40;
            af[mt].q[0] = *(const u32x4*)&al[base + 16 * hi];
            af[mt].q[1] = *(const u32x4*)&al[base + 16 * hi + 8];
        }
#pragma unroll
        for (int nt = 0; nt < 4; ++nt) {
            int base = (wn + nt * 16 + lm) * 40 + 16 * hi;
            bf[nt].q[0] = *(const u32x4*)&bl[base];
            bf[nt].q[1] = *(const u32x4*)&bl[base + 8];
        }
#pragma unroll
        for (int mt = 0; mt < 2; ++mt)
#pragma unroll
            for (int nt = 0; nt < 4; ++nt)
                acc[mt][nt] = WMMA_BF16(af[mt].v, bf[nt].v, acc[mt][nt]);
    }
#undef GEMM_STAGE

    // Epilogue: D layout -> VGPR r holds row (r + 8*hi), col (lane&15)
#pragma unroll
    for (int mt = 0; mt < 2; ++mt)
#pragma unroll
        for (int nt = 0; nt < 4; ++nt) {
            int n = n0 + wn + nt * 16 + lm;
            float bv = bias[n];
#pragma unroll
            for (int r = 0; r < 8; ++r) {
                int m = m0 + wm + mt * 16 + r + 8 * hi;
                float val = acc[mt][nt][r] + bv;
                if (MODE == 0) {
                    int b = m >> 10, t = m & 1023, h = n >> 6, d = n & 63;
                    ((__bf16*)outp)[(((b * 16 + h) * 1024) + t) * 64 + d] =
                        (__bf16)(val * oscale);
                } else {
                    ((float*)outp)[m * 1024 + n] = val;
                }
            }
        }
}

// ---------------------------------------------------------------------------
// nROPE over Q and K, head layout [B*H][T][64]; rotate-half pairs (32+j,48+j).
// (Q carries the folded 1/sqrt(HD) scale; rotation is linear so it commutes.)
// ---------------------------------------------------------------------------
__global__ void rope_kernel(__bf16* __restrict__ Q, __bf16* __restrict__ K,
                            const float* __restrict__ nf) {
    int tid = blockIdx.x * 256 + threadIdx.x;
    int j  = tid & 15;
    int t  = (tid >> 4) & 1023;
    int bh = tid >> 14;

    float sig  = 1.0f / (1.0f + __expf(-nf[0]));
    float base = sig * 3400.0f + 8300.0f;               // SDT_L, MIN_L*BASE
    float invf = __powf(base, -(float)j * (1.0f / 16.0f));
    float ang  = (float)t * invf * (6.283185307179586f / 1023.0f);
    float c = __cosf(ang), s = __sinf(ang);

    long idx = ((long)bh * 1024 + t) * 64;
    float q1 = (float)Q[idx + 32 + j], q2 = (float)Q[idx + 48 + j];
    Q[idx + 32 + j] = (__bf16)(q1 * c - q2 * s);
    Q[idx + 48 + j] = (__bf16)(q1 * s + q2 * c);
    float k1 = (float)K[idx + 32 + j], k2 = (float)K[idx + 48 + j];
    K[idx + 32 + j] = (__bf16)(k1 * c - k2 * s);
    K[idx + 48 + j] = (__bf16)(k1 * s + k2 * c);
}

// ---------------------------------------------------------------------------
// Flash attention, one workgroup per (b,h, 128-row q block), 8 waves.
// K tiles: async double-buffered LDS. V tiles: b128 register prefetch,
// scattered to transposed LDS. Online softmax: asm permlane16 max butterfly,
// lane-partial exp sums (reduced once at the end). Q carries 1/sqrt(HD).
// ---------------------------------------------------------------------------
__global__ __launch_bounds__(256, 1)
void attn_kernel(const __bf16* __restrict__ Q, const __bf16* __restrict__ K,
                 const __bf16* __restrict__ V, __bf16* __restrict__ AO) {
    __shared__ __align__(16) __bf16 kbuf[2][32 * 72];     // [key][d]  pad 72
    __shared__ __align__(16) __bf16 vbuf[2][64 * 40];     // [d][key]  pad 40
    __shared__ __align__(16) __bf16 pbuf[8 * 16 * 40];    // per-wave [m][key]

    const int tid  = threadIdx.x;
    const int lane = tid & 31;
    const int w    = tid >> 5;
    const int hi   = lane >> 4;
    const int lm   = lane & 15;

    const int qblk = blockIdx.x;          // 0..7
    const int bh   = blockIdx.y;          // 0..63
    const int b    = bh >> 4, h = bh & 15;

    const __bf16* Qh = Q + (long)bh * 1024 * 64;
    const __bf16* Kh = K + (long)bh * 1024 * 64;
    const __bf16* Vh = V + (long)bh * 1024 * 64;

    // Wave's Q A-operand fragments (d chunks 0..31 and 32..63), loaded once.
    const int qrow = qblk * 128 + w * 16 + lm;
    const __bf16* qp = Qh + qrow * 64;
    Frag16 aq0, aq1;
    aq0.q[0] = *(const u32x4*)(qp + 16 * hi);
    aq0.q[1] = *(const u32x4*)(qp + 16 * hi + 8);
    aq1.q[0] = *(const u32x4*)(qp + 32 + 16 * hi);
    aq1.q[1] = *(const u32x4*)(qp + 40 + 16 * hi);

    float rmax[8], rsum[8];               // rsum is a per-lane partial!
    v8f o[4];
#pragma unroll
    for (int r = 0; r < 8; ++r) { rmax[r] = -1e30f; rsum[r] = 0.0f; }
#pragma unroll
    for (int f = 0; f < 4; ++f) o[f] = {};

    __bf16* pb = &pbuf[w * 16 * 40];

    // staging geometry: K tile = 256 x 16B segments; V tile = 1 x 16B /thread
    const int krow = tid >> 3, kc8 = (tid & 7) * 8;
    const int vkey = tid >> 3, vdc = (tid & 7) * 8;

    VReg vcur, vnxt;
    CP_ASYNC_B128(&kbuf[0][krow * 72 + kc8], &Kh[krow * 64 + kc8]);
    vcur.q = *(const u32x4*)&Vh[vkey * 64 + vdc];

    for (int ti = 0; ti < 32; ++ti) {
        const int p  = ti & 1;
        const int kt = ti * 32;
        CP_WAIT();
        // scatter V registers into transposed LDS tile vbuf[p][d][key]
        __bf16* vb = vbuf[p];
#pragma unroll
        for (int e = 0; e < 8; ++e) vb[(vdc + e) * 40 + vkey] = vcur.h[e];
        __syncthreads();

        if (ti + 1 < 32) {
            CP_ASYNC_B128(&kbuf[1 - p][krow * 72 + kc8], &Kh[(kt + 32 + krow) * 64 + kc8]);
            vnxt.q = *(const u32x4*)&Vh[(kt + 32 + vkey) * 64 + vdc];
        }

        const __bf16* kb = kbuf[p];

        // Scores: S[16 q][32 keys] as two 16x16 f32 tiles, contraction d=0..63
        v8f s0 = {}, s1 = {};
        {
            Frag16 bk;
            bk.q[0] = *(const u32x4*)&kb[lm * 72 + 16 * hi];
            bk.q[1] = *(const u32x4*)&kb[lm * 72 + 16 * hi + 8];
            s0 = WMMA_BF16(aq0.v, bk.v, s0);
            bk.q[0] = *(const u32x4*)&kb[lm * 72 + 32 + 16 * hi];
            bk.q[1] = *(const u32x4*)&kb[lm * 72 + 32 + 16 * hi + 8];
            s0 = WMMA_BF16(aq1.v, bk.v, s0);
            bk.q[0] = *(const u32x4*)&kb[(16 + lm) * 72 + 16 * hi];
            bk.q[1] = *(const u32x4*)&kb[(16 + lm) * 72 + 16 * hi + 8];
            s1 = WMMA_BF16(aq0.v, bk.v, s1);
            bk.q[0] = *(const u32x4*)&kb[(16 + lm) * 72 + 32 + 16 * hi];
            bk.q[1] = *(const u32x4*)&kb[(16 + lm) * 72 + 32 + 16 * hi + 8];
            s1 = WMMA_BF16(aq1.v, bk.v, s1);
        }

        // Online softmax per q-row (row r+8*hi lives in 16 lanes of a half).
        // Row max must be lane-uniform (feeds P scaling); exp-sum stays
        // lane-partial and is reduced once after the loop.
#pragma unroll
        for (int r = 0; r < 8; ++r) {
            float a0 = s0[r];
            float a1 = s1[r];
            float tm = red_max16_fast(fmax_fast(a0, a1));
            float mnew = fmax_fast(rmax[r], tm);
            float corr = __expf(rmax[r] - mnew);
            float p0 = __expf(a0 - mnew);
            float p1 = __expf(a1 - mnew);
            rsum[r] = rsum[r] * corr + (p0 + p1);
            rmax[r] = mnew;
#pragma unroll
            for (int f = 0; f < 4; ++f) o[f][r] *= corr;
            int m = r + 8 * hi;
            pb[m * 40 + lm]      = (__bf16)p0;
            pb[m * 40 + 16 + lm] = (__bf16)p1;
        }
        asm volatile("s_wait_dscnt 0" ::: "memory");   // wave-local pbuf RAW

        // O[16][64] += P[16][32] @ V[32][64]
        Frag16 ap;
        ap.q[0] = *(const u32x4*)&pb[lm * 40 + 16 * hi];
        ap.q[1] = *(const u32x4*)&pb[lm * 40 + 16 * hi + 8];
        const __bf16* vbl = vbuf[p];
#pragma unroll
        for (int f = 0; f < 4; ++f) {
            Frag16 bv;
            int rb = (f * 16 + lm) * 40 + 16 * hi;
            bv.q[0] = *(const u32x4*)&vbl[rb];
            bv.q[1] = *(const u32x4*)&vbl[rb + 8];
            o[f] = WMMA_BF16(ap.v, bv.v, o[f]);
        }
        vcur = vnxt;
    }

    // Normalize (reduce the lane-partial sums now) and write [B*T][E] bf16.
    const int tbase = qblk * 128 + w * 16;
#pragma unroll
    for (int r = 0; r < 8; ++r) {
        float tot = red_sum16_fast(rsum[r]);
        float inv = 1.0f / tot;
        int m = r + 8 * hi;
        long row = (long)b * 1024 + tbase + m;
#pragma unroll
        for (int f = 0; f < 4; ++f)
            AO[row * 1024 + h * 64 + f * 16 + lm] = (__bf16)(o[f][r] * inv);
    }
}

// ---------------------------------------------------------------------------
// Host-side orchestration.
// inputs: x, Wq, bq, Wk, bk, Wv, bv, Wo, bo, nrope_freq
// ---------------------------------------------------------------------------
extern "C" void kernel_launch(void* const* d_in, const int* in_sizes, int n_in,
                              void* d_out, int out_size, void* d_ws, size_t ws_size,
                              hipStream_t stream) {
    (void)in_sizes; (void)n_in; (void)out_size; (void)ws_size;
    const float* x  = (const float*)d_in[0];
    const float* Wq = (const float*)d_in[1];
    const float* bq = (const float*)d_in[2];
    const float* Wk = (const float*)d_in[3];
    const float* bk = (const float*)d_in[4];
    const float* Wv = (const float*)d_in[5];
    const float* bv = (const float*)d_in[6];
    const float* Wo = (const float*)d_in[7];
    const float* bo = (const float*)d_in[8];
    const float* nf = (const float*)d_in[9];

    char* ws = (char*)d_ws;
    const size_t MB = 1ull << 20;
    __bf16* xb  = (__bf16*)(ws);             // 8 MiB   [4096][1024]
    __bf16* AO  = xb;                        // aliased: xb dead after QKV GEMMs
    __bf16* wqt = (__bf16*)(ws + 8 * MB);    // 2 MiB each, [N][K] transposed
    __bf16* wkt = (__bf16*)(ws + 10 * MB);
    __bf16* wvt = (__bf16*)(ws + 12 * MB);
    __bf16* wot = (__bf16*)(ws + 14 * MB);
    __bf16* Qb  = (__bf16*)(ws + 16 * MB);   // 8 MiB each, [B*H][T][64]
    __bf16* Kb  = (__bf16*)(ws + 24 * MB);
    __bf16* Vb  = (__bf16*)(ws + 32 * MB);

    convert_x_kernel<<<4096, 256, 0, stream>>>(x, xb);
    transpose_w_kernel<<<dim3(32, 32, 4), dim3(32, 32), 0, stream>>>(
        Wq, Wk, Wv, Wo, wqt, wkt, wvt, wot);

    // Q carries 1/sqrt(HD) = 0.125
    gemm_bf16_kernel<0><<<dim3(8, 32), 256, 0, stream>>>(xb, wqt, bq, (void*)Qb, 0.125f);
    gemm_bf16_kernel<0><<<dim3(8, 32), 256, 0, stream>>>(xb, wkt, bk, (void*)Kb, 1.0f);
    gemm_bf16_kernel<0><<<dim3(8, 32), 256, 0, stream>>>(xb, wvt, bv, (void*)Vb, 1.0f);

    rope_kernel<<<4096, 256, 0, stream>>>(Qb, Kb, nf);

    attn_kernel<<<dim3(8, 64), 256, 0, stream>>>(Qb, Kb, Vb, AO);

    gemm_bf16_kernel<1><<<dim3(8, 32), 256, 0, stream>>>(AO, wot, bo, d_out, 1.0f);
}
